// SelfAttention_47665547051099
// MI455X (gfx1250) — matrix operand order repaired and checked by reading, not compiled, not run on hardware
//
#include <hip/hip_runtime.h>
#include <math.h>

typedef __attribute__((ext_vector_type(16))) _Float16 v16h;
typedef __attribute__((ext_vector_type(8)))  _Float16 v8h;
typedef __attribute__((ext_vector_type(8)))  float    v8f;
typedef int v4i_vs __attribute__((vector_size(16)));   // matches builtin param type

#define SS   4096   // tokens (64*64)
#define CC   512    // channels
#define NH   8      // heads
#define HD   64     // head dim

#define ROPE_LOG_MIN 1.14472988584940017f   // ln(pi)
#define ROPE_STEP    0.03597789281444906f   // ln(10)/64

#define HAS_ASYNC_LDS __has_builtin(__builtin_amdgcn_global_load_async_to_lds_b128)

static __device__ __forceinline__ v8f wmma_f16(v16h a, v16h b, v8f c) {
  return __builtin_amdgcn_wmma_f32_16x16x32_f16(false, a, false, b, (short)0, c,
                                                false, false);
}

// Assemble an A-operand (16x32 f16) from 16 contiguous halves (k = 16*half + s).
static __device__ __forceinline__ v16h load_a_op(const _Float16* p, int half) {
  v8h lo = *(const v8h*)(p + 16 * half);
  v8h hi = *(const v8h*)(p + 16 * half + 8);
  v16h a;
#pragma unroll
  for (int i = 0; i < 8; i++) { a[i] = lo[i]; a[i + 8] = hi[i]; }
  return a;
}

// Assemble a B-operand from 16 contiguous halves via two 16B loads (16B aligned).
static __device__ __forceinline__ v16h load_b_op(const _Float16* p) {
  v8h lo = *(const v8h*)(p);
  v8h hi = *(const v8h*)(p + 8);
  v16h b;
#pragma unroll
  for (int i = 0; i < 8; i++) { b[i] = lo[i]; b[i + 8] = hi[i]; }
  return b;
}

// 16-byte global -> LDS copy: async on CDNA5 (ASYNCcnt), fallback = load+store.
static __device__ __forceinline__ void copy16_g2l(const _Float16* g, _Float16* l) {
#if HAS_ASYNC_LDS
  __builtin_amdgcn_global_load_async_to_lds_b128(
      (__attribute__((address_space(1))) v4i_vs*)g,
      (__attribute__((address_space(3))) v4i_vs*)l, 0, 0);
#else
  *(v8h*)l = *(const v8h*)g;
#endif
}

template <int N> static __device__ __forceinline__ void wait_async() {
#if HAS_ASYNC_LDS
#if __has_builtin(__builtin_amdgcn_s_wait_asynccnt)
  __builtin_amdgcn_s_wait_asynccnt(N);
#else
  asm volatile("s_wait_asynccnt %0" ::"n"(N) : "memory");
#endif
#endif
}

// ---------------------------------------------------------------- prep kernels
// Transpose + convert X[c][s] f32 -> Xt[s][c] f16 via padded LDS tile
// (both global phases fully coalesced; LDS row stride 68 halves = conflict-free).
__global__ __launch_bounds__(256) void cvt_transpose_x(const float* __restrict__ X,
                                                       _Float16* __restrict__ Xt) {
  __shared__ _Float16 tile[64][68];
  int t = threadIdx.x;
  int s0 = (blockIdx.x & 63) * 64;   // 64 s-tiles
  int c0 = (blockIdx.x >> 6) * 64;   // 8 c-tiles
  int sl = t & 63, c4 = t >> 6;
#pragma unroll
  for (int p = 0; p < 16; p++) {
    int cl = c4 + p * 4;
    tile[sl][cl] = (_Float16)X[(c0 + cl) * SS + s0 + sl];
  }
  __syncthreads();
  int cl = t & 63, s4 = t >> 6;
#pragma unroll
  for (int p = 0; p < 16; p++) {
    int srow = s4 + p * 4;
    Xt[(s0 + srow) * CC + c0 + cl] = tile[srow][cl];
  }
}

__global__ __launch_bounds__(256) void cvt_f16(const float* __restrict__ in,
                                               _Float16* __restrict__ out, int nel) {
  int i = blockIdx.x * 256 + threadIdx.x;
  if (i < nel) out[i] = (_Float16)in[i];
}

// ---------------------------------------------------------------- WMMA GEMM
// C[M x 4096] = A[M x 512] * B^T, B stored [4096 x 512] (N-major, K contiguous).
// Block = 8 waves: shared 64-col B tile, per-wave 16-row A tile -> 128x64 output.
// K stepped by 32 with a 4-deep async global->LDS pipeline (staged 3 ahead).
#define NBUF 4
#define LROW 40   // LDS row stride in halves (padding: 20*n mod 64 permutes banks)

__global__ __launch_bounds__(256) void gemm_wmma_f16(
    const _Float16* __restrict__ A, const _Float16* __restrict__ B,
    float* __restrict__ Cout, const float* __restrict__ bias, int M) {
  __shared__ __align__(64) _Float16 ldsB[NBUF][64 * LROW];      // 20 KB
  __shared__ __align__(64) _Float16 ldsA[NBUF][8][16 * LROW];   // 40 KB
  int tid = threadIdx.x;
  int w = tid >> 5, lane = tid & 31;
  int n = lane & 15, half = lane >> 4;
  int mblk = blockIdx.x >> 6;                    // 64 n-groups of width 64
  int nb = (blockIdx.x & 63) * 64;
  int mt = mblk * 8 + w;                         // wave's 16-row m-tile

  const _Float16* Arow0 = A + mt * 16 * CC;

  // Stage one k-step into buffer bi: 3 async instructions per wave.
  auto stage = [&](int k0, int bi) {
    {  // shared B tile: 64 rows x 32 halves, one 16B chunk per thread
      int r = tid >> 2, c = tid & 3;
      copy16_g2l(B + (nb + r) * CC + k0 + c * 8, &ldsB[bi][r * LROW + c * 8]);
    }
#pragma unroll
    for (int i = 0; i < 2; i++) {  // private A tile: 16 rows x 32 halves
      int idx = i * 32 + lane;
      int r = idx >> 2, c = idx & 3;
      copy16_g2l(Arow0 + r * CC + k0 + c * 8, &ldsA[bi][w][r * LROW + c * 8]);
    }
  };

  v8f acc[4] = {};

  auto compute = [&](int t) {
    int bi = t & (NBUF - 1);
    v16h a = load_a_op(&ldsA[bi][w][n * LROW], half);
#pragma unroll
    for (int j = 0; j < 4; j++) {
      v16h b = load_b_op(&ldsB[bi][(j * 16 + n) * LROW + 16 * half]);
      acc[j] = wmma_f16(a, b, acc[j]);
    }
  };

  stage(0, 0);
  stage(32, 1);
  stage(64, 2);
  // 16 k-steps total; steady state keeps 2 future buffers (6 copies) in flight.
#pragma unroll 1
  for (int t = 0; t < 13; t++) {
    wait_async<6>();     // buffer t complete (own copies; barrier covers others)
    __syncthreads();     // also proves everyone consumed buffer t-1
    stage((t + 3) * 32, (t + 3) & (NBUF - 1));
    compute(t);
  }
  wait_async<6>(); __syncthreads(); compute(13);
  wait_async<3>(); __syncthreads(); compute(14);
  wait_async<0>(); __syncthreads(); compute(15);

#pragma unroll
  for (int r = 0; r < 8; r++) {
    int row = mt * 16 + r + 8 * half;            // C/D: M = r + 8*half
    float bv = bias ? bias[row] : 0.0f;
#pragma unroll
    for (int j = 0; j < 4; j++)
      Cout[row * SS + nb + j * 16 + n] = acc[j][r] + bv;
  }
}

// ---------------------------------------------------------------- depthwise 3x3
__global__ __launch_bounds__(256) void dwconv3x3(const float* __restrict__ in,
                                                 const float* __restrict__ w,
                                                 const float* __restrict__ b,
                                                 float* __restrict__ out) {
  int idx = blockIdx.x * 256 + threadIdx.x;        // 1536*4096
  int o = idx >> 12, s = idx & 4095;
  int y = s >> 6, x = s & 63;
  const float* base = in + o * SS;
  const float* wp = w + o * 9;
  float acc = b[o];
#pragma unroll
  for (int dy = -1; dy <= 1; dy++)
#pragma unroll
    for (int dx = -1; dx <= 1; dx++) {
      int yy = y + dy, xx = x + dx;
      if ((unsigned)yy < 64u && (unsigned)xx < 64u)
        acc += wp[(dy + 1) * 3 + (dx + 1)] * base[yy * 64 + xx];
    }
  out[idx] = acc;
}

// ---------------------------------------------------------------- RoPE + pack
// qh[h][s][d] (q pre-scaled by 1/8), kh[h][s][d], vt[h][d][s]  -- all f16.
// Block = one head x 64 tokens. q/k results staged in padded LDS tiles so the
// d-contiguous global writes are coalesced; vt writes are s-contiguous already.
__global__ __launch_bounds__(256) void rope_pack(const float* __restrict__ qkv,
                                                 const float* __restrict__ pos,
                                                 _Float16* __restrict__ qh,
                                                 _Float16* __restrict__ kh,
                                                 _Float16* __restrict__ vt) {
  __shared__ _Float16 qtile[64][68];
  __shared__ _Float16 ktile[64][68];
  int t = threadIdx.x;
  int h = blockIdx.x >> 6;
  int s0 = (blockIdx.x & 63) * 64;
  int sl = t & 63;            // token within tile (coalesced loads)
  int jj = t >> 6;            // 0..3 (uniform per wave)
  int s = s0 + sl;
  const float* qp = qkv + (h * HD) * SS + s;
  const float* kp = qkv + (CC + h * HD) * SS + s;
  const float* vp = qkv + (2 * CC + h * HD) * SS + s;
  _Float16* vo = vt + h * HD * SS + s;
  float py = pos[2 * s], px = pos[2 * s + 1];
#pragma unroll
  for (int p = 0; p < 8; p++) {
    int j = jj * 8 + p;       // 0..31
    if (j < 16) {
      int d1 = j, d2 = j + 16;
      int fidx = (j & 7) * 8 + h;
      float pc = (j < 8) ? py : px;
      float th = pc * __expf(ROPE_LOG_MIN + (float)fidx * ROPE_STEP);
      float cs, sn;
      __sincosf(th, &sn, &cs);
      float q1 = qp[d1 * SS], q2 = qp[d2 * SS];
      float k1 = kp[d1 * SS], k2 = kp[d2 * SS];
      qtile[sl][d1] = (_Float16)(0.125f * (q1 * cs - q2 * sn));
      qtile[sl][d2] = (_Float16)(0.125f * (q2 * cs + q1 * sn));
      ktile[sl][d1] = (_Float16)(k1 * cs - k2 * sn);
      ktile[sl][d2] = (_Float16)(k2 * cs + k1 * sn);
      vo[d1 * SS] = (_Float16)vp[d1 * SS];
      vo[d2 * SS] = (_Float16)vp[d2 * SS];
    } else {
      int d1 = 16 + j, d2 = 32 + j;   // 32..47, 48..63 pass-through
      qtile[sl][d1] = (_Float16)(0.125f * qp[d1 * SS]);
      qtile[sl][d2] = (_Float16)(0.125f * qp[d2 * SS]);
      ktile[sl][d1] = (_Float16)kp[d1 * SS];
      ktile[sl][d2] = (_Float16)kp[d2 * SS];
      vo[d1 * SS] = (_Float16)vp[d1 * SS];
      vo[d2 * SS] = (_Float16)vp[d2 * SS];
    }
  }
  __syncthreads();
  int d = t & 63, r0 = t >> 6;
#pragma unroll
  for (int p = 0; p < 16; p++) {
    int srow = r0 + p * 4;
    qh[(h * SS + s0 + srow) * HD + d] = qtile[srow][d];
    kh[(h * SS + s0 + srow) * HD + d] = ktile[srow][d];
  }
}

// ---------------------------------------------------------------- flash attention
// One wave: one head, 16 query rows, streams 4096 keys in chunks of 32.
// K tiles double-buffered in registers; V tiles issued before the softmax so
// their latency hides behind the VALU section. Output aoT[s][c] f16 (c=h*64+d).
__global__ __launch_bounds__(256) void attn_flash(const _Float16* __restrict__ qh,
                                                  const _Float16* __restrict__ kh,
                                                  const _Float16* __restrict__ vt,
                                                  _Float16* __restrict__ aoT) {
  __shared__ __align__(32) _Float16 lds_p[8][16 * 32];  // per-wave P staging
  int wave = (blockIdx.x * 256 + threadIdx.x) >> 5;
  int wl = threadIdx.x >> 5;
  int lane = threadIdx.x & 31;
  int n = lane & 15, half = lane >> 4;
  int h = wave >> 8;
  int q0 = (wave & 255) * 16;

  const _Float16* qp = qh + h * SS * HD;
  const _Float16* kp = kh + h * SS * HD;
  const _Float16* vp = vt + h * HD * SS;

  // Q tile in A-operand layout: a0 = dims 0..31, a1 = dims 32..63
  const _Float16* qrow = qp + (q0 + n) * HD;
  v16h a0 = load_a_op(qrow, half);
  v16h a1 = load_a_op(qrow + 32, half);

  v8f acc[4] = {};
  float mrow[8], lrow[8];
#pragma unroll
  for (int r = 0; r < 8; r++) { mrow[r] = -3.0e38f; lrow[r] = 0.0f; }

  _Float16* pl = lds_p[wl];

  v16h kb[4], kn[4];
  {
    const _Float16* kr0 = kp + n * HD + 16 * half;
    kb[0] = *(const v16h*)(kr0);
    kb[1] = *(const v16h*)(kr0 + 32);
    const _Float16* kr1 = kp + (16 + n) * HD + 16 * half;
    kb[2] = *(const v16h*)(kr1);
    kb[3] = *(const v16h*)(kr1 + 32);
  }

  auto chunk = [&](int kc, bool prefetch) {
    // V tiles for this chunk: issued first, consumed last
    v16h bv[4];
#pragma unroll
    for (int j = 0; j < 4; j++)
      bv[j] = *(const v16h*)(vp + (j * 16 + n) * SS + kc + 16 * half);
    // K tiles for next chunk
    if (prefetch) {
      const _Float16* kr0 = kp + (kc + 32 + n) * HD + 16 * half;
      kn[0] = *(const v16h*)(kr0);
      kn[1] = *(const v16h*)(kr0 + 32);
      const _Float16* kr1 = kp + (kc + 48 + n) * HD + 16 * half;
      kn[2] = *(const v16h*)(kr1);
      kn[3] = *(const v16h*)(kr1 + 32);
    }
    // scores: two 16-key groups, contraction over d via two chained WMMAs each
    v8f s0 = {}, s1 = {};
    s0 = wmma_f16(a0, kb[0], s0);
    s0 = wmma_f16(a1, kb[1], s0);
    s1 = wmma_f16(a0, kb[2], s1);
    s1 = wmma_f16(a1, kb[3], s1);
    // online softmax per query row (rows live in 16-lane groups)
#pragma unroll
    for (int r = 0; r < 8; r++) {
      float v0 = s0[r], v1 = s1[r];
      float mx = fmaxf(v0, v1);
#pragma unroll
      for (int off = 1; off < 16; off <<= 1) mx = fmaxf(mx, __shfl_xor(mx, off, 32));
      float mnew = fmaxf(mrow[r], mx);
      float corr = __expf(mrow[r] - mnew);
      float p0 = __expf(v0 - mnew);
      float p1 = __expf(v1 - mnew);
      float sum = p0 + p1;
#pragma unroll
      for (int off = 1; off < 16; off <<= 1) sum += __shfl_xor(sum, off, 32);
      lrow[r] = lrow[r] * corr + sum;
      mrow[r] = mnew;
      acc[0][r] *= corr; acc[1][r] *= corr; acc[2][r] *= corr; acc[3][r] *= corr;
      int row = r + 8 * half;
      pl[row * 32 + n] = (_Float16)p0;        // C-layout -> plain [row][key] in LDS
      pl[row * 32 + 16 + n] = (_Float16)p1;
    }
    __asm__ volatile("s_wait_dscnt 0" ::: "memory");
    // reload P in A-operand layout (same-wave DS ordering guarantees visibility)
    v16h pa = load_a_op(pl + n * 32, half);
    // O += P @ V : contraction over 32 keys, 4 d-groups of 16
#pragma unroll
    for (int j = 0; j < 4; j++) acc[j] = wmma_f16(pa, bv[j], acc[j]);
    if (prefetch) {
#pragma unroll
      for (int j = 0; j < 4; j++) kb[j] = kn[j];
    }
  };

#pragma unroll 1
  for (int kc = 0; kc < SS - 32; kc += 32) chunk(kc, true);
  chunk(SS - 32, false);

  // finalize: divide by l, write aoT[s][c]
#pragma unroll
  for (int r = 0; r < 8; r++) {
    float inv = 1.0f / lrow[r];
    int srow = q0 + r + 8 * half;
#pragma unroll
    for (int j = 0; j < 4; j++)
      aoT[srow * CC + h * HD + j * 16 + n] = (_Float16)(acc[j][r] * inv);
  }
}

// ---------------------------------------------------------------- launch
extern "C" void kernel_launch(void* const* d_in, const int* in_sizes, int n_in,
                              void* d_out, int out_size, void* d_ws, size_t ws_size,
                              hipStream_t stream) {
  const float* x     = (const float*)d_in[0];   // [512,4096]
  const float* pos   = (const float*)d_in[1];   // [4096,2]
  const float* w_qkv = (const float*)d_in[2];   // [1536,512]
  const float* w_dw  = (const float*)d_in[3];   // [1536,9]
  const float* b_dw  = (const float*)d_in[4];   // [1536]
  const float* w_out = (const float*)d_in[5];   // [512,512]
  const float* b_out = (const float*)d_in[6];   // [512]
  float* out = (float*)d_out;                   // [512,4096]

  char* ws = (char*)d_ws;
  size_t off = 0;
  auto alloc = [&](size_t bytes) { char* p = ws + off; off += (bytes + 255) & ~(size_t)255; return p; };
  _Float16* Xt      = (_Float16*)alloc((size_t)SS * CC * 2);       // [s][c]
  _Float16* wqh     = (_Float16*)alloc((size_t)3 * CC * CC * 2);   // [o][c]
  _Float16* woh     = (_Float16*)alloc((size_t)CC * CC * 2);       // [o][c]
  float*    qkv_raw = (float*)   alloc((size_t)3 * CC * SS * 4);   // [o][s]
  float*    qkv_dw  = (float*)   alloc((size_t)3 * CC * SS * 4);   // [o][s]
  _Float16* qh      = (_Float16*)alloc((size_t)NH * SS * HD * 2);  // [h][s][d]
  _Float16* kh      = (_Float16*)alloc((size_t)NH * SS * HD * 2);  // [h][s][d]
  _Float16* vt      = (_Float16*)alloc((size_t)NH * HD * SS * 2);  // [h][d][s]
  _Float16* aoT     = (_Float16*)alloc((size_t)SS * CC * 2);       // [s][c]
  (void)ws_size; (void)in_sizes; (void)n_in; (void)out_size;

  // prep: transpose+convert activations, convert weights to f16
  cvt_transpose_x<<<8 * 64, 256, 0, stream>>>(x, Xt);
  cvt_f16<<<(3 * CC * CC) / 256, 256, 0, stream>>>(w_qkv, wqh, 3 * CC * CC);
  cvt_f16<<<(CC * CC) / 256, 256, 0, stream>>>(w_out, woh, CC * CC);

  // QKV projection: M=1536 -> (1536/128) m-blocks * 64 n-groups = 768 blocks
  gemm_wmma_f16<<<(1536 / 128) * 64, 256, 0, stream>>>(wqh, Xt, qkv_raw, nullptr, 3 * CC);

  // depthwise 3x3 + bias
  dwconv3x3<<<(3 * CC * SS) / 256, 256, 0, stream>>>(qkv_raw, w_dw, b_dw, qkv_dw);

  // RoPE + pack to f16 attention layouts: 8 heads x 64 token-tiles
  rope_pack<<<NH * 64, 256, 0, stream>>>(qkv_dw, pos, qh, kh, vt);

  // flash attention: 8 heads * 256 q-tiles = 2048 waves
  attn_flash<<<(NH * 256) / 8, 256, 0, stream>>>(qh, kh, vt, aoT);

  // output projection + bias: M=512 -> (512/128) * 64 = 256 blocks
  gemm_wmma_f16<<<(512 / 128) * 64, 256, 0, stream>>>(woh, aoT, out, b_out, CC);
}
